// MultiHeadAttention_23416161698056
// MI455X (gfx1250) — matrix operand order repaired and checked by reading, not compiled, not run on hardware
//
#include <hip/hip_runtime.h>

// ---------------------------------------------------------------------------
// MultiHeadAttention forward, MI455X (gfx1250), wave32 + WMMA f16->f32,
// async global->LDS staging with double buffering.
// B=2, T=2048, D=1024, H=16, DH=64.
// ---------------------------------------------------------------------------

#define BB 2
#define TT 2048
#define DD 1024
#define HH 16
#define DHH 64
#define MM (BB * TT) // 4096 rows

typedef __attribute__((ext_vector_type(16))) _Float16 v16h;
typedef __attribute__((ext_vector_type(8)))  _Float16 v8h;
typedef __attribute__((ext_vector_type(8)))  float    v8f;
typedef __attribute__((ext_vector_type(4)))  int      v4i;

union AFrag { v16h v; v8h h[2]; };
union Acc   { v8f  v; float f[8]; };

// ---- CDNA5 async global->LDS copy (ASYNCcnt tracked), with fallback --------
#if defined(__HIP_DEVICE_COMPILE__) && \
    __has_builtin(__builtin_amdgcn_global_load_async_to_lds_b128)
#define HAVE_ASYNC_LDS 1
#else
#define HAVE_ASYNC_LDS 0
#endif

static __device__ __forceinline__ void async_cp16(const _Float16* g,
                                                  _Float16* l) {
#if HAVE_ASYNC_LDS
  __builtin_amdgcn_global_load_async_to_lds_b128(
      (__attribute__((address_space(1))) v4i*)(g),
      (__attribute__((address_space(3))) v4i*)(l), 0, 0);
#else
  *(v8h*)l = *(const v8h*)g;
#endif
}

static __device__ __forceinline__ void wait_async_all() {
#if HAVE_ASYNC_LDS
#if __has_builtin(__builtin_amdgcn_s_wait_asynccnt)
  __builtin_amdgcn_s_wait_asynccnt(0);
#else
  asm volatile("s_wait_asynccnt 0x0" ::: "memory");
#endif
#endif
}

static __device__ __forceinline__ v8h cvt_f32x8_to_h8(float4 a, float4 b) {
  v8h r;
  r[0] = (_Float16)a.x; r[1] = (_Float16)a.y; r[2] = (_Float16)a.z; r[3] = (_Float16)a.w;
  r[4] = (_Float16)b.x; r[5] = (_Float16)b.y; r[6] = (_Float16)b.z; r[7] = (_Float16)b.w;
  return r;
}

// ---------------------------------------------------------------------------
// Prep 1: x (f32) -> xh (f16), row major [4096][1024].
// ---------------------------------------------------------------------------
__global__ __launch_bounds__(256) void cvt_x_f16(const float* __restrict__ x,
                                                 _Float16* __restrict__ xh) {
  const size_t i = ((size_t)blockIdx.x * 256 + threadIdx.x) * 8;
  const float4* s = (const float4*)(x + i);
  *(v8h*)(xh + i) = cvt_f32x8_to_h8(s[0], s[1]);
}

// ---------------------------------------------------------------------------
// Prep 2: W (f32, [k][n]) -> Wt (f16, [n][k]) for Wq/Wk/Wv/Wo (blockIdx.z).
// One-time transpose so GEMM B tiles become contiguous async copies.
// ---------------------------------------------------------------------------
__global__ __launch_bounds__(256) void cvt_w_transpose(
    const float* __restrict__ Wq, const float* __restrict__ Wk,
    const float* __restrict__ Wv, const float* __restrict__ Wo,
    _Float16* __restrict__ Wt4) {
  const int z = blockIdx.z;
  const float* W = (z == 0) ? Wq : (z == 1) ? Wk : (z == 2) ? Wv : Wo;
  _Float16* Wt = Wt4 + (size_t)z * DD * DD;
  const size_t idx = ((size_t)blockIdx.x * 256 + threadIdx.x) * 8;
  const int k = (int)(idx >> 10);
  const int n = (int)(idx & (DD - 1));
  const float4* s = (const float4*)(W + idx);
  float4 a = s[0], b = s[1];
  float vv[8] = {a.x, a.y, a.z, a.w, b.x, b.y, b.z, b.w};
#pragma unroll
  for (int q = 0; q < 8; q++) Wt[(size_t)(n + q) * DD + k] = (_Float16)vv[q];
}

// ---------------------------------------------------------------------------
// Kernel 1: QKV projection.  C = xh @ W, f16 out.
// Block 256 thr (8 waves); tile M=128 x N=64, K-step 64, double buffered.
// Wave tile 16x64 -> 4 accumulators, 8 WMMA / k-iter.
// Q scaled by 1/sqrt(DH).  Q/K out: [bh][t][dh].  V out transposed: [bh][dh][t].
// ---------------------------------------------------------------------------
__global__ __launch_bounds__(256) void qkv_gemm_wmma(
    const _Float16* __restrict__ xh, const _Float16* __restrict__ Wt4,
    _Float16* __restrict__ q_ws, _Float16* __restrict__ k_ws,
    _Float16* __restrict__ vt_ws) {
  __shared__ __align__(16) _Float16 lA[2][128 * 64];  // [m][k]
  __shared__ __align__(16) _Float16 lB[2][64 * 64];   // [n][k]

  const int tid = threadIdx.x, wave = tid >> 5, lane = tid & 31, ln = lane & 15;
  const int n0 = blockIdx.x * 64, m0 = blockIdx.y * 128, which = blockIdx.z;
  const _Float16* Wt = Wt4 + (size_t)which * DD * DD;
  const float scale = (which == 0) ? 0.125f : 1.0f;

  const int ar = tid >> 1, ac = (tid & 1) * 32;  // A staging: 32 halves/thread
  const int br = tid >> 2, bc = (tid & 3) * 16;  // B staging: 16 halves/thread

  Acc acc[4];
#pragma unroll
  for (int j = 0; j < 4; j++) acc[j].v = (v8f)0.0f;

  auto stage = [&](int k0, int buf) {
#pragma unroll
    for (int u = 0; u < 4; u++)
      async_cp16(xh + (size_t)(m0 + ar) * DD + k0 + ac + u * 8,
                 &lA[buf][ar * 64 + ac + u * 8]);
#pragma unroll
    for (int u = 0; u < 2; u++)
      async_cp16(Wt + (size_t)(n0 + br) * DD + k0 + bc + u * 8,
                 &lB[buf][br * 64 + bc + u * 8]);
  };

  stage(0, 0);
  wait_async_all();
  __syncthreads();

  const int arow = wave * 16 + ln;
  const int kk8 = (lane < 16) ? 0 : 8;
  const int kk16 = (lane < 16) ? 0 : 16;

  for (int kt = 0; kt < DD / 64; kt++) {
    const int cur = kt & 1;
    if (kt + 1 < DD / 64) stage((kt + 1) * 64, 1 - cur);

    AFrag af[2];
#pragma unroll
    for (int f = 0; f < 2; f++) {
      af[f].h[0] = *(const v8h*)(&lA[cur][arow * 64 + f * 32 + kk8]);
      af[f].h[1] = *(const v8h*)(&lA[cur][arow * 64 + f * 32 + kk8 + 16]);
    }
#pragma unroll
    for (int j = 0; j < 4; j++) {
#pragma unroll
      for (int f = 0; f < 2; f++) {
        AFrag bf;
        bf.h[0] = *(const v8h*)(&lB[cur][(j * 16 + ln) * 64 + f * 32 + kk8]);
        bf.h[1] = *(const v8h*)(&lB[cur][(j * 16 + ln) * 64 + f * 32 + kk8 + 16]);
        acc[j].v = __builtin_amdgcn_wmma_f32_16x16x32_f16(
            false, af[f].v, false, bf.v, (short)0, acc[j].v, false, false);
      }
    }
    wait_async_all();
    __syncthreads();
  }

#pragma unroll
  for (int j = 0; j < 4; j++) {
#pragma unroll
    for (int i = 0; i < 8; i++) {
      const int row = m0 + wave * 16 + ((lane < 16) ? i : 8 + i);
      const int col = n0 + j * 16 + ln;
      const int b = row >> 11, t = row & (TT - 1);
      const int h = col >> 6, d = col & 63;
      const _Float16 hv = (_Float16)(acc[j].f[i] * scale);
      if (which == 2)
        vt_ws[(((size_t)(b * HH + h)) * DHH + d) * TT + t] = hv;
      else {
        _Float16* dst = which ? k_ws : q_ws;
        dst[(((size_t)(b * HH + h)) * TT + t) * DHH + d] = hv;
      }
    }
  }
}

// ---------------------------------------------------------------------------
// Kernel 2: causal flash attention.  Grid (T/128, B*H).  8 waves x 16 q-rows.
// 64-key tiles, double-buffered async staging, online softmax.
// 16 WMMA per k-iter per wave (8 for S, 8 for PV).
// ---------------------------------------------------------------------------
__global__ __launch_bounds__(256) void flash_attn_wmma(
    const _Float16* __restrict__ q_ws, const _Float16* __restrict__ k_ws,
    const _Float16* __restrict__ vt_ws, _Float16* __restrict__ ctx_ws) {
  __shared__ __align__(16) _Float16 lK[2][64 * 64];   // [key][dh]
  __shared__ __align__(16) _Float16 lVt[2][64 * 64];  // [dh][key]
  __shared__ __align__(16) _Float16 lP[8][16 * 64];   // per-wave P scratch

  const int tid = threadIdx.x, wave = tid >> 5, lane = tid & 31, ln = lane & 15;
  const int bh = blockIdx.y;
  const int blk0 = blockIdx.x * 128;
  const int q0w = blk0 + wave * 16;

  const _Float16* Qh = q_ws + (size_t)bh * TT * DHH;
  const _Float16* Kh = k_ws + (size_t)bh * TT * DHH;
  const _Float16* Vt = vt_ws + (size_t)bh * DHH * TT;  // [dh][T]

  const int kk8 = (lane < 16) ? 0 : 8;
  const int kk16 = (lane < 16) ? 0 : 16;

  // Q resident in registers: two 16x32 A-fragments
  AFrag qa[2];
  {
    const int row = q0w + ln;
#pragma unroll
    for (int f = 0; f < 2; f++) {
      qa[f].h[0] = *(const v8h*)(Qh + (size_t)row * DHH + f * 32 + kk8);
      qa[f].h[1] = *(const v8h*)(Qh + (size_t)row * DHH + f * 32 + kk8 + 16);
    }
  }

  Acc c[4];
#pragma unroll
  for (int t = 0; t < 4; t++) c[t].v = (v8f)0.0f;
  float mrow[8], lrow[8];
#pragma unroll
  for (int i = 0; i < 8; i++) { mrow[i] = -1e30f; lrow[i] = 0.0f; }

  // staging: 16 halves per thread for K and for V
  const int s_r = (tid * 16) >> 6;  // 0..63
  const int s_c = (tid * 16) & 63;  // 0,16,32,48
  auto stage = [&](int kt, int buf) {
    const int k0 = kt * 64;
#pragma unroll
    for (int u = 0; u < 2; u++)
      async_cp16(Kh + (size_t)(k0 + s_r) * DHH + s_c + u * 8,
                 &lK[buf][s_r * 64 + s_c + u * 8]);
#pragma unroll
    for (int u = 0; u < 2; u++)
      async_cp16(Vt + (size_t)s_r * TT + k0 + s_c + u * 8,
                 &lVt[buf][s_r * 64 + s_c + u * 8]);
  };

  const int nkt = (blk0 >> 6) + 2;
  stage(0, 0);
  wait_async_all();
  __syncthreads();

  for (int kt = 0; kt < nkt; kt++) {
    const int cur = kt & 1;
    const int k0 = kt * 64;
    if (kt + 1 < nkt) stage(kt + 1, 1 - cur);

    if (k0 <= q0w + 15) {  // wave-uniform causal skip
      // ---- S = Q @ K^T : 16 x 64 scores ----
      Acc s[4];
#pragma unroll
      for (int j = 0; j < 4; j++) s[j].v = (v8f)0.0f;
#pragma unroll
      for (int j = 0; j < 4; j++) {
#pragma unroll
        for (int f = 0; f < 2; f++) {
          AFrag bf;  // lane = key col, halves = dh kk8..+7 and kk8+16..+23
          bf.h[0] = *(const v8h*)(&lK[cur][(j * 16 + ln) * 64 + f * 32 + kk8]);
          bf.h[1] = *(const v8h*)(&lK[cur][(j * 16 + ln) * 64 + f * 32 + kk8 + 16]);
          s[j].v = __builtin_amdgcn_wmma_f32_16x16x32_f16(
              false, qa[f].v, false, bf.v, (short)0, s[j].v, false, false);
        }
      }
      // ---- causal mask ----
#pragma unroll
      for (int j = 0; j < 4; j++) {
        const int key = k0 + j * 16 + ln;
#pragma unroll
        for (int i = 0; i < 8; i++) {
          const int qrow = q0w + ((lane < 16) ? i : 8 + i);
          if (key > qrow) s[j].f[i] = -1e30f;
        }
      }
      // ---- online softmax ----
#pragma unroll
      for (int i = 0; i < 8; i++) {
        float r = fmaxf(fmaxf(s[0].f[i], s[1].f[i]),
                        fmaxf(s[2].f[i], s[3].f[i]));
#pragma unroll
        for (int off = 1; off < 16; off <<= 1)
          r = fmaxf(r, __shfl_xor(r, off, 32));
        const float mnew = fmaxf(mrow[i], r);
        const float alpha = __expf(mrow[i] - mnew);
        float rs = 0.0f;
#pragma unroll
        for (int j = 0; j < 4; j++) {
          const float p = __expf(s[j].f[i] - mnew);
          s[j].f[i] = p;
          rs += p;
        }
#pragma unroll
        for (int off = 1; off < 16; off <<= 1)
          rs += __shfl_xor(rs, off, 32);
        lrow[i] = lrow[i] * alpha + rs;
        mrow[i] = mnew;
        c[0].f[i] *= alpha; c[1].f[i] *= alpha;
        c[2].f[i] *= alpha; c[3].f[i] *= alpha;
      }
      // ---- P (C layout) -> A layout via per-wave LDS ----
      _Float16* pw = lP[wave];
#pragma unroll
      for (int j = 0; j < 4; j++)
#pragma unroll
        for (int i = 0; i < 8; i++) {
          const int row = (lane < 16) ? i : 8 + i;
          pw[row * 64 + j * 16 + ln] = (_Float16)s[j].f[i];
        }
      AFrag pf[2];
#pragma unroll
      for (int c2 = 0; c2 < 2; c2++) {
        pf[c2].h[0] = *(const v8h*)(pw + ln * 64 + c2 * 32 + kk8);
        pf[c2].h[1] = *(const v8h*)(pw + ln * 64 + c2 * 32 + kk8 + 16);
      }
      // ---- ctx += P @ V ----
#pragma unroll
      for (int t = 0; t < 4; t++) {
#pragma unroll
        for (int c2 = 0; c2 < 2; c2++) {
          AFrag vf;  // lane = dh col, halves = keys kk8..+7 and kk8+16..+23
          bool lo = (lane < 16);
          vf.h[0] = *(const v8h*)(&lVt[cur][(t * 16 + ln) * 64 + c2 * 32 + (lo ? 0 : 8)]);
          vf.h[1] = *(const v8h*)(&lVt[cur][(t * 16 + ln) * 64 + c2 * 32 + (lo ? 0 : 8) + 16]);
          c[t].v = __builtin_amdgcn_wmma_f32_16x16x32_f16(
              false, pf[c2].v, false, vf.v, (short)0, c[t].v, false, false);
        }
      }
    }
    wait_async_all();
    __syncthreads();
  }

  // ---- epilogue: ctx /= l, store f16 [b*T + t][h*64 + dh] ----
  const int b = bh / HH, h = bh % HH;
#pragma unroll
  for (int t = 0; t < 4; t++) {
#pragma unroll
    for (int i = 0; i < 8; i++) {
      const int qrow = q0w + ((lane < 16) ? i : 8 + i);
      const int col = h * DHH + t * 16 + ln;
      const float inv = 1.0f / lrow[i];
      ctx_ws[((size_t)(b * TT + qrow)) * DD + col] = (_Float16)(c[t].f[i] * inv);
    }
  }
}

// ---------------------------------------------------------------------------
// Kernel 3: output projection.  out(f32) = ctx(f16) @ Wo + bo.
// Same tiling as kernel 1; B from pre-transposed Wot (f16).
// ---------------------------------------------------------------------------
__global__ __launch_bounds__(256) void out_gemm_wmma(
    const _Float16* __restrict__ ctx_ws, const _Float16* __restrict__ Wot,
    const float* __restrict__ bo, float* __restrict__ out) {
  __shared__ __align__(16) _Float16 lA[2][128 * 64];
  __shared__ __align__(16) _Float16 lB[2][64 * 64];

  const int tid = threadIdx.x, wave = tid >> 5, lane = tid & 31, ln = lane & 15;
  const int n0 = blockIdx.x * 64, m0 = blockIdx.y * 128;

  const int ar = tid >> 1, ac = (tid & 1) * 32;
  const int br = tid >> 2, bc = (tid & 3) * 16;

  Acc acc[4];
#pragma unroll
  for (int j = 0; j < 4; j++) acc[j].v = (v8f)0.0f;

  auto stage = [&](int k0, int buf) {
#pragma unroll
    for (int u = 0; u < 4; u++)
      async_cp16(ctx_ws + (size_t)(m0 + ar) * DD + k0 + ac + u * 8,
                 &lA[buf][ar * 64 + ac + u * 8]);
#pragma unroll
    for (int u = 0; u < 2; u++)
      async_cp16(Wot + (size_t)(n0 + br) * DD + k0 + bc + u * 8,
                 &lB[buf][br * 64 + bc + u * 8]);
  };

  stage(0, 0);
  wait_async_all();
  __syncthreads();

  const int arow = wave * 16 + ln;
  const int kk8 = (lane < 16) ? 0 : 8;
  const int kk16 = (lane < 16) ? 0 : 16;

  for (int kt = 0; kt < DD / 64; kt++) {
    const int cur = kt & 1;
    if (kt + 1 < DD / 64) stage((kt + 1) * 64, 1 - cur);

    AFrag af[2];
#pragma unroll
    for (int f = 0; f < 2; f++) {
      af[f].h[0] = *(const v8h*)(&lA[cur][arow * 64 + f * 32 + kk8]);
      af[f].h[1] = *(const v8h*)(&lA[cur][arow * 64 + f * 32 + kk8 + 16]);
    }
#pragma unroll
    for (int j = 0; j < 4; j++) {
#pragma unroll
      for (int f = 0; f < 2; f++) {
        AFrag bf;
        bf.h[0] = *(const v8h*)(&lB[cur][(j * 16 + ln) * 64 + f * 32 + kk8]);
        bf.h[1] = *(const v8h*)(&lB[cur][(j * 16 + ln) * 64 + f * 32 + kk8 + 16]);
        acc[j].v = __builtin_amdgcn_wmma_f32_16x16x32_f16(
            false, af[f].v, false, bf.v, (short)0, acc[j].v, false, false);
      }
    }
    wait_async_all();
    __syncthreads();
  }

#pragma unroll
  for (int j = 0; j < 4; j++) {
#pragma unroll
    for (int i = 0; i < 8; i++) {
      const int row = m0 + wave * 16 + ((lane < 16) ? i : 8 + i);
      const int col = n0 + j * 16 + ln;
      out[(size_t)row * DD + col] = acc[j].f[i] + bo[col];
    }
  }
}

// ---------------------------------------------------------------------------
// Host side
// ---------------------------------------------------------------------------
extern "C" void kernel_launch(void* const* d_in, const int* in_sizes, int n_in,
                              void* d_out, int out_size, void* d_ws,
                              size_t ws_size, hipStream_t stream) {
  (void)in_sizes; (void)n_in; (void)out_size; (void)ws_size;
  const float* x = (const float*)d_in[0];
  const float* Wq = (const float*)d_in[1];
  const float* Wk = (const float*)d_in[2];
  const float* Wv = (const float*)d_in[3];
  const float* Wo = (const float*)d_in[4];
  const float* bo = (const float*)d_in[5];
  float* out = (float*)d_out;

  char* ws = (char*)d_ws;
  _Float16* xh    = (_Float16*)(ws);                       // 8 MB
  _Float16* Wt4   = (_Float16*)(ws + ((size_t)8 << 20));   // 8 MB (4 x 2MB)
  _Float16* q_ws  = (_Float16*)(ws + ((size_t)16 << 20));  // 8 MB
  _Float16* k_ws  = (_Float16*)(ws + ((size_t)24 << 20));  // 8 MB
  _Float16* vt_ws = (_Float16*)(ws + ((size_t)32 << 20));  // 8 MB (transposed)
  _Float16* ctx   = (_Float16*)(ws + ((size_t)40 << 20));  // 8 MB

  cvt_x_f16<<<(MM * DD / 8) / 256, 256, 0, stream>>>(x, xh);
  cvt_w_transpose<<<dim3((DD * DD / 8) / 256, 1, 4), 256, 0, stream>>>(
      Wq, Wk, Wv, Wo, Wt4);
  qkv_gemm_wmma<<<dim3(DD / 64, MM / 128, 3), 256, 0, stream>>>(
      xh, Wt4, q_ws, k_ws, vt_ws);
  flash_attn_wmma<<<dim3(TT / 128, BB * HH), 256, 0, stream>>>(
      q_ws, k_ws, vt_ws, ctx);
  out_gemm_wmma<<<dim3(DD / 64, MM / 128), 256, 0, stream>>>(
      ctx, Wt4 + (size_t)3 * DD * DD, bo, out);
}
